// InteractiveNet_42099269435678
// MI455X (gfx1250) — matrix operand order repaired and checked by reading, not compiled, not run on hardware
//
#include <hip/hip_runtime.h>

#define BN 4
#define NN 384
#define DD 64

typedef __bf16 bf16_t;
typedef __bf16 v16bf __attribute__((ext_vector_type(16)));
typedef __bf16 v8bf  __attribute__((ext_vector_type(8)));
typedef float  v8f   __attribute__((ext_vector_type(8)));

#define EDGE_STRIDE 264   // 16x256 tile, padded rows (528B) -> conflict-free A loads
#define TILE_STRIDE 72    // 16x64 tiles, padded rows (144B)

__device__ __forceinline__ float fast_sigmoid(float x) {
    return __builtin_amdgcn_rcpf(1.0f + __expf(-x));   // v_exp_f32 + v_rcp_f32
}
__device__ __forceinline__ float fast_silu(float x) { return x * fast_sigmoid(x); }

__device__ __forceinline__ v8f zero8() {
    v8f z;
#pragma unroll
    for (int t = 0; t < 8; ++t) z[t] = 0.0f;
    return z;
}

// A fragment (16x32 bf16, this lane's share): p = tile + M*stride + kbase.
// ISA layout: lanes 0-15 M=0..15; VGPR0..3 K=hi*8+0..7, VGPR4..7 K=16+hi*8+0..7
// -> two contiguous 16B LDS reads (ds_load_b128) per fragment.
__device__ __forceinline__ v16bf load_a_frag(const bf16_t* p, int hi) {
    const v8bf lo = *(const v8bf*)(p + hi * 8);
    const v8bf hv = *(const v8bf*)(p + 16 + hi * 8);
    v16bf a;
#pragma unroll
    for (int t = 0; t < 8; ++t) { a[t] = lo[t]; a[t + 8] = hv[t]; }
    return a;
}

// B fragment (32x16 bf16) built once per block from global f32 weights W[K][64].
// Lane holds column n; K = kbase + hi*8 + t (t<8), kbase + 16 + hi*8 + (t-8) (t>=8). Half-wave reads 64B contiguous per t.
__device__ __forceinline__ v16bf load_b_frag(const float* __restrict__ W, int kbase, int hi, int n) {
    v16bf b;
#pragma unroll
    for (int t = 0; t < 16; ++t) b[t] = (bf16_t)W[(kbase + hi * 8 + (t & 7) + (t >> 3) * 16) * DD + n];
    return b;
}

__global__ __launch_bounds__(128)
void InteractiveNet_fused_kernel(const float* __restrict__ h, const float* __restrict__ e,
                                 const float* __restrict__ W1, const float* __restrict__ b1,
                                 const float* __restrict__ W2, const float* __restrict__ b2,
                                 const float* __restrict__ Wg, const float* __restrict__ bg,
                                 const float* __restrict__ Wm, const float* __restrict__ bm,
                                 const float* __restrict__ Wu, const float* __restrict__ bu,
                                 const float* __restrict__ Wq, const float* __restrict__ bq,
                                 float* __restrict__ out) {
    __shared__ float  s_ef [16 * DD];            // e tile, f32 (exact blend)
    __shared__ __align__(16) bf16_t s_edge[16 * EDGE_STRIDE]; // [hs+ht,|hs-ht|,hs*ht,e] bf16
    __shared__ __align__(16) bf16_t s_hsb [16 * TILE_STRIDE]; // hs bf16 (msg A, ks 0-1)
    __shared__ __align__(16) bf16_t s_t1  [16 * TILE_STRIDE]; // silu(edge_in@W1) bf16
    __shared__ __align__(16) bf16_t s_de  [16 * TILE_STRIDE]; // de bf16 (gate A, ks 2-3)
    __shared__ __align__(16) bf16_t s_en  [16 * TILE_STRIDE]; // e_new bf16 (msg A, ks 4-5)
    __shared__ __align__(16) bf16_t s_htb [DD];  // h_target bf16 (row-broadcast)
    __shared__ float s_htf[DD];
    __shared__ float s_b1[DD], s_b2[DD], s_bg[DD], s_bm[DD];
    __shared__ float s_agg[DD];
    __shared__ float s_dh[DD];

    const int tid  = threadIdx.x;
    const int lane = tid & 31;
    const int wv   = tid >> 5;
    const int hi   = lane >> 4;
    const int ln   = lane & 15;          // A-row / B-column index within 16
    const int n    = wv * 16 + ln;       // output column owned by this lane
    const int blk  = blockIdx.x;
    const int bb   = blk / NN;
    const int ii   = blk % NN;

    if (tid < DD) {
        s_b1[tid] = b1[tid]; s_b2[tid] = b2[tid];
        s_bg[tid] = bg[tid]; s_bm[tid] = bm[tid];
        const float hv = h[(bb * NN + ii) * DD + tid];
        s_htf[tid] = hv; s_htb[tid] = (bf16_t)hv;
        s_agg[tid] = 0.0f;
    }

    // --- weight B-fragments resident in VGPRs for the whole block ---
    v16bf B1f[8], B2f[2], Bgf[4], Bmf[6];
#pragma unroll
    for (int ks = 0; ks < 8; ++ks) B1f[ks] = load_b_frag(W1, 32 * ks, hi, n);
#pragma unroll
    for (int ks = 0; ks < 2; ++ks) B2f[ks] = load_b_frag(W2, 32 * ks, hi, n);
#pragma unroll
    for (int ks = 0; ks < 4; ++ks) Bgf[ks] = load_b_frag(Wg, 32 * ks, hi, n);
#pragma unroll
    for (int ks = 0; ks < 6; ++ks) Bmf[ks] = load_b_frag(Wm, 32 * ks, hi, n);
    __syncthreads();

    float* outH = out;
    float* outE = out + (size_t)BN * NN * DD;
    const float* hb = h + (size_t)bb * NN * DD;
    const float* eb = e + (size_t)(bb * NN + ii) * NN * DD;

    // software-pipelined tile staging: next tile lives in registers during GEMMs
    const int sr = tid >> 6;            // this thread's base row (0/1), rows step by 2
    const int sc = tid & 63;            // this thread's column
    float rh[8], re[8];

#pragma unroll
    for (int q = 0; q < 8; ++q) {       // preload tile j0 = 0
        const int r = sr + 2 * q;
        rh[q] = hb[r * DD + sc];
        re[q] = eb[r * DD + sc];
    }

#pragma unroll 1
    for (int j0 = 0; j0 < NN; j0 += 16) {
        // ---- drain registers -> LDS, fusing edge_in = [hs+ht,|hs-ht|,hs*ht,e] ----
        {
            const float ht = s_htf[sc];
#pragma unroll
            for (int q = 0; q < 8; ++q) {
                const int r = sr + 2 * q;
                const float hs = rh[q], ev = re[q];
                bf16_t* er = s_edge + r * EDGE_STRIDE + sc;
                er[0]   = (bf16_t)(hs + ht);
                er[64]  = (bf16_t)fabsf(hs - ht);
                er[128] = (bf16_t)(hs * ht);
                er[192] = (bf16_t)ev;
                s_hsb[r * TILE_STRIDE + sc] = (bf16_t)hs;
                s_ef[r * DD + sc] = ev;
            }
        }
        __syncthreads();

        // ---- issue next tile's global loads; they overlap the 4 GEMM stages ----
        if (j0 + 16 < NN) {
#pragma unroll
            for (int q = 0; q < 8; ++q) {
                const int r = j0 + 16 + sr + 2 * q;
                rh[q] = hb[r * DD + sc];
                re[q] = eb[r * DD + sc];
            }
        }

        // ---- GEMM1: t1 = silu(edge_in @ W_eu1 + b1), K=256 ----
        {
            v8f acc = zero8();
            const bf16_t* Ar = s_edge + ln * EDGE_STRIDE;
#pragma unroll
            for (int ks = 0; ks < 8; ++ks) {
                const v16bf a = load_a_frag(Ar + 32 * ks, hi);
                acc = __builtin_amdgcn_wmma_f32_16x16x32_bf16(false, a, false, B1f[ks],
                                                              (short)0, acc, false, false);
            }
            const float bias = s_b1[n];
#pragma unroll
            for (int r = 0; r < 8; ++r) {
                const float x = fast_silu(acc[r] + bias);
                s_t1[(r + hi * 8) * TILE_STRIDE + n] = (bf16_t)x;
            }
        }
        __syncthreads();

        // ---- GEMM2: de = t1 @ W_eu2 + b2, K=64 ----
        float de[8];
        {
            v8f acc = zero8();
            const bf16_t* Ar = s_t1 + ln * TILE_STRIDE;
#pragma unroll
            for (int ks = 0; ks < 2; ++ks) {
                const v16bf a = load_a_frag(Ar + 32 * ks, hi);
                acc = __builtin_amdgcn_wmma_f32_16x16x32_bf16(false, a, false, B2f[ks],
                                                              (short)0, acc, false, false);
            }
            const float bias = s_b2[n];
#pragma unroll
            for (int r = 0; r < 8; ++r) {
                de[r] = acc[r] + bias;
                s_de[(r + hi * 8) * TILE_STRIDE + n] = (bf16_t)de[r];
            }
        }
        __syncthreads();

        // ---- GEMM3: eg = sigmoid([e, de] @ W_eg + bg); e_new = eg*e + (1-eg)*de ----
        {
            v8f acc = zero8();
            acc = __builtin_amdgcn_wmma_f32_16x16x32_bf16(false,
                      load_a_frag(s_edge + ln * EDGE_STRIDE + 192, hi),      false, Bgf[0], (short)0, acc, false, false);
            acc = __builtin_amdgcn_wmma_f32_16x16x32_bf16(false,
                      load_a_frag(s_edge + ln * EDGE_STRIDE + 192 + 32, hi), false, Bgf[1], (short)0, acc, false, false);
            acc = __builtin_amdgcn_wmma_f32_16x16x32_bf16(false,
                      load_a_frag(s_de + ln * TILE_STRIDE, hi),              false, Bgf[2], (short)0, acc, false, false);
            acc = __builtin_amdgcn_wmma_f32_16x16x32_bf16(false,
                      load_a_frag(s_de + ln * TILE_STRIDE + 32, hi),         false, Bgf[3], (short)0, acc, false, false);
            const float bias = s_bg[n];
#pragma unroll
            for (int r = 0; r < 8; ++r) {
                const int M = r + hi * 8;
                const float g  = fast_sigmoid(acc[r] + bias);
                const float ef = s_ef[M * DD + n];
                const float en = g * ef + (1.0f - g) * de[r];
                s_en[M * TILE_STRIDE + n] = (bf16_t)en;
                outE[((size_t)blk * NN + (j0 + M)) * DD + n] = en;
            }
        }
        __syncthreads();

        // ---- GEMM4: msg = silu([hs, ht, e_new] @ W_msg + bm); masked row-sum ----
        {
            v8f acc = zero8();
            acc = __builtin_amdgcn_wmma_f32_16x16x32_bf16(false,
                      load_a_frag(s_hsb + ln * TILE_STRIDE, hi),      false, Bmf[0], (short)0, acc, false, false);
            acc = __builtin_amdgcn_wmma_f32_16x16x32_bf16(false,
                      load_a_frag(s_hsb + ln * TILE_STRIDE + 32, hi), false, Bmf[1], (short)0, acc, false, false);
            acc = __builtin_amdgcn_wmma_f32_16x16x32_bf16(false,
                      load_a_frag(s_htb, hi),                         false, Bmf[2], (short)0, acc, false, false);
            acc = __builtin_amdgcn_wmma_f32_16x16x32_bf16(false,
                      load_a_frag(s_htb + 32, hi),                    false, Bmf[3], (short)0, acc, false, false);
            acc = __builtin_amdgcn_wmma_f32_16x16x32_bf16(false,
                      load_a_frag(s_en + ln * TILE_STRIDE, hi),       false, Bmf[4], (short)0, acc, false, false);
            acc = __builtin_amdgcn_wmma_f32_16x16x32_bf16(false,
                      load_a_frag(s_en + ln * TILE_STRIDE + 32, hi),  false, Bmf[5], (short)0, acc, false, false);
            const float bias = s_bm[n];
            float part = 0.0f;
#pragma unroll
            for (int r = 0; r < 8; ++r) {
                const int j = j0 + r + hi * 8;
                const float mv = fast_silu(acc[r] + bias);
                if (j != ii) part += mv;       // diagonal mask
            }
            atomicAdd(&s_agg[n], part);
        }
        __syncthreads();
    }

    // ---- node update (tiny M=1 GEMVs) ----
    float dh = 0.0f, hval = 0.0f;
    if (tid < DD) {
        const float inv = 1.0f / (float)(NN - 1);
        hval = s_htf[tid];
        float acc = bu[tid];
#pragma unroll 4
        for (int k = 0; k < DD; ++k) acc += s_htf[k] * Wu[k * DD + tid];
#pragma unroll 4
        for (int k = 0; k < DD; ++k) acc += (s_agg[k] * inv) * Wu[(DD + k) * DD + tid];
        dh = fast_silu(acc);
        s_dh[tid] = dh;
    }
    __syncthreads();
    if (tid < DD) {
        float acc = bq[tid];
#pragma unroll 4
        for (int k = 0; k < DD; ++k) acc += s_htf[k] * Wq[k * DD + tid];
#pragma unroll 4
        for (int k = 0; k < DD; ++k) acc += s_dh[k] * Wq[(DD + k) * DD + tid];
        const float g = fast_sigmoid(acc);
        outH[(size_t)blk * DD + tid] = g * hval + (1.0f - g) * dh;
    }
}

extern "C" void kernel_launch(void* const* d_in, const int* in_sizes, int n_in,
                              void* d_out, int out_size, void* d_ws, size_t ws_size,
                              hipStream_t stream) {
    const float* h  = (const float*)d_in[0];
    const float* e  = (const float*)d_in[1];
    const float* W1 = (const float*)d_in[2];  const float* b1 = (const float*)d_in[3];
    const float* W2 = (const float*)d_in[4];  const float* b2 = (const float*)d_in[5];
    const float* Wg = (const float*)d_in[6];  const float* bg = (const float*)d_in[7];
    const float* Wm = (const float*)d_in[8];  const float* bm = (const float*)d_in[9];
    const float* Wu = (const float*)d_in[10]; const float* bu = (const float*)d_in[11];
    const float* Wq = (const float*)d_in[12]; const float* bq = (const float*)d_in[13];

    dim3 grid(BN * NN), block(128);
    hipLaunchKernelGGL(InteractiveNet_fused_kernel, grid, block, 0, stream,
                       h, e, W1, b1, W2, b2, Wg, bg, Wm, bm, Wu, bu, Wq, bq,
                       (float*)d_out);
}
